// GraphAutoencoder_42271068127545
// MI455X (gfx1250) — matrix operand order repaired and checked by reading, not compiled, not run on hardware
//
#include <hip/hip_runtime.h>
#include <hip/hip_bf16.h>
#include <math.h>

// ---------------- problem constants (match reference) ----------------
#define GN 32768      // nodes
#define GE 524288     // edges
#define GT 4096       // tokens
#define GB 64         // graphs
#define GNP 512       // nodes per graph
#define GDE 128
#define GDH 256
#define GDL 64

typedef __attribute__((ext_vector_type(16))) __bf16 v16bf;
typedef __attribute__((ext_vector_type(8)))  __bf16 v8bf;
typedef __attribute__((ext_vector_type(4)))  __bf16 bf16x4;
typedef __attribute__((ext_vector_type(8)))  float  v8f;
typedef __attribute__((ext_vector_type(4)))  float  f32x4;

// ---------------- utility kernels ----------------
__global__ void zero_f32_kernel(float* p, long long n) {
    long long i = (long long)blockIdx.x * blockDim.x + threadIdx.x;
    if (i < n) p[i] = 0.0f;
}
__global__ void zero_u32_kernel(unsigned* p, long long n) {
    long long i = (long long)blockIdx.x * blockDim.x + threadIdx.x;
    if (i < n) p[i] = 0u;
}

// count occurrences: cnt[idx[i]] += 1
__global__ void count_u32_kernel(const int* __restrict__ idx, unsigned* __restrict__ cnt, int n) {
    int i = blockIdx.x * blockDim.x + threadIdx.x;
    if (i < n) atomicAdd(&cnt[idx[i]], 1u);
}

// deg = indeg + 1 (self loop); dis = deg^-1/2; self_norm = 1/deg
__global__ void node_norm_kernel(const unsigned* __restrict__ indeg,
                                 float* __restrict__ dis, float* __restrict__ self_norm, int n) {
    int i = blockIdx.x * blockDim.x + threadIdx.x;
    if (i < n) {
        float deg = (float)indeg[i] + 1.0f;
        dis[i] = rsqrtf(deg);
        self_norm[i] = 1.0f / deg;
    }
}

// single-block chunked exclusive scan
__global__ void exclusive_scan_u32_kernel(const unsigned* __restrict__ in,
                                          unsigned* __restrict__ out, int n) {
    __shared__ unsigned temp[256];
    __shared__ unsigned carry;
    if (threadIdx.x == 0) carry = 0u;
    __syncthreads();
    for (int base = 0; base < n; base += 256) {
        int i = base + (int)threadIdx.x;
        unsigned v = (i < n) ? in[i] : 0u;
        temp[threadIdx.x] = v;
        __syncthreads();
        for (int off = 1; off < 256; off <<= 1) {
            unsigned t = (threadIdx.x >= (unsigned)off) ? temp[threadIdx.x - off] : 0u;
            __syncthreads();
            temp[threadIdx.x] += t;
            __syncthreads();
        }
        if (i < n) out[i] = carry + temp[threadIdx.x] - v;  // exclusive
        __syncthreads();
        if (threadIdx.x == 255) carry += temp[255];
        __syncthreads();
    }
}

// build CSR sorted-by-dst: csr_src / csr_w (= dis[src]*dis[dst])
__global__ void csr_fill_kernel(const int* __restrict__ src, const int* __restrict__ dst,
                                const float* __restrict__ dis,
                                const unsigned* __restrict__ offsets, unsigned* __restrict__ cursor,
                                unsigned* __restrict__ csr_src, float* __restrict__ csr_w, int e) {
    int i = blockIdx.x * blockDim.x + threadIdx.x;
    if (i >= e) return;
    int s = src[i], d = dst[i];
    unsigned slot = atomicAdd(&cursor[d], 1u);
    unsigned idx = offsets[d] + slot;
    csr_src[idx] = (unsigned)s;
    csr_w[idx] = dis[s] * dis[d];
}

// h0[i][:] = emb[x[i]][:]  (float4 vectorized, 32 threads/node)
__global__ void gather_emb_kernel(const int* __restrict__ x, const float* __restrict__ emb,
                                  float* __restrict__ h0) {
    int i = blockIdx.x;
    int c = threadIdx.x;            // blockDim == GDE/4 == 32
    ((f32x4*)h0)[(size_t)i * (GDE / 4) + c] =
        ((const f32x4*)emb)[(size_t)x[i] * (GDE / 4) + c];
}

// out[i][:] = relu?( sum_j w_j * hw[src_j][:] + self_norm[i]*hw[i][:] + bias[:] )
// float4 per thread; blockDim == Dq = D/4
__global__ void gcn_aggregate_kernel(const float* __restrict__ hw, float* __restrict__ out,
                                     const unsigned* __restrict__ csr_src,
                                     const float* __restrict__ csr_w,
                                     const unsigned* __restrict__ offsets,
                                     const unsigned* __restrict__ indeg,
                                     const float* __restrict__ self_norm,
                                     const float* __restrict__ bias,
                                     int Dq, int relu) {
    int node = blockIdx.x;
    int c = threadIdx.x;            // blockDim == Dq
    const f32x4* hwv = (const f32x4*)hw;
    f32x4 acc = hwv[(size_t)node * Dq + c] * self_norm[node] + ((const f32x4*)bias)[c];
    unsigned s = offsets[node];
    unsigned e = s + indeg[node];
    for (unsigned j = s; j < e; ++j) {
        unsigned sn = csr_src[j];
        if (j + 1u < e)             // gfx1250 global_prefetch_b8 for next gathered row
            __builtin_prefetch(&hwv[(size_t)csr_src[j + 1u] * Dq + c], 0, 0);
        acc += hwv[(size_t)sn * Dq + c] * csr_w[j];
    }
    if (relu) {
        acc.x = fmaxf(acc.x, 0.0f); acc.y = fmaxf(acc.y, 0.0f);
        acc.z = fmaxf(acc.z, 0.0f); acc.w = fmaxf(acc.w, 0.0f);
    }
    ((f32x4*)out)[(size_t)node * Dq + c] = acc;
}

// graph mean pool: accumulate then divide
__global__ void pool_accum_kernel(const float* __restrict__ z, const int* __restrict__ batch,
                                  float* __restrict__ ge) {
    int i = blockIdx.x;
    int c = threadIdx.x;            // blockDim == GDL
    atomicAdd(&ge[(size_t)batch[i] * GDL + c], z[(size_t)i * GDL + c]);
}
__global__ void pool_div_kernel(float* __restrict__ ge, const unsigned* __restrict__ counts) {
    int b = blockIdx.x;
    int c = threadIdx.x;
    ge[(size_t)b * GDL + c] /= (float)counts[b];
}

// to_dense_batch scatter (+ mask as 1.0f); float4 vectorized, 16 threads/node
__global__ void scatter_dense_kernel(const float* __restrict__ z, const int* __restrict__ batch,
                                     const unsigned* __restrict__ starts,
                                     float* __restrict__ z_dense, float* __restrict__ mask) {
    int i = blockIdx.x;
    int c = threadIdx.x;            // blockDim == GDL/4 == 16
    int b = batch[i];
    int pos = i - (int)starts[b];
    ((f32x4*)z_dense)[((size_t)b * GNP + pos) * (GDL / 4) + c] =
        ((const f32x4*)z)[(size_t)i * (GDL / 4) + c];
    if (c == 0) mask[(size_t)b * GNP + pos] = 1.0f;
}

// ---------------- WMMA GEMM: C = A[M,K] x B + epilogue ----------------
// btrans==0: B is [K][Nn] row-major.  btrans==1: B is [Nn][K] row-major (C = A Bt^T).
// mode: 0 = none, 1 = +bias, 3 = sigmoid.  Batched via blockIdx.z with element strides.
// Requires M%128==0, Nn%64==0, K%32==0 (true for all call sites).
__device__ inline v16bf mk_frag(v8bf lo, v8bf hi) {
    v16bf r;
#pragma unroll
    for (int i = 0; i < 8; ++i) { r[i] = lo[i]; r[i + 8] = hi[i]; }
    return r;
}

__global__ void wmma_gemm_kernel(const float* __restrict__ A, const float* __restrict__ B,
                                 const float* __restrict__ bias, float* __restrict__ C,
                                 int M, int Nn, int K, int btrans, int mode,
                                 long long sA, long long sB, long long sC) {
    A += (long long)blockIdx.z * sA;
    B += (long long)blockIdx.z * sB;
    C += (long long)blockIdx.z * sC;

    const int m0 = blockIdx.x * 128;
    const int n0 = blockIdx.y * 64;

    __shared__ __align__(16) __bf16 As[128][40];  // [m][k], 80B row stride
    __shared__ __align__(16) __bf16 Bs[64][40];   // [n][k] (B transposed in LDS)

    const int tid  = threadIdx.x;
    const int lane = tid & 31;
    const int wv   = tid >> 5;                    // 8 waves
    const int half = lane >> 4;
    const int lr   = lane & 15;

    v8f zero = {};
    v8f acc0 = zero, acc1 = zero, acc2 = zero, acc3 = zero;

    for (int kt = 0; kt < K; kt += 32) {
        // ---- stage A tile: 128x32, global_load_b128 -> v_cvt_pk_bf16_f32 x2 -> ds_store_b64 ----
#pragma unroll
        for (int u = 0; u < 4; ++u) {
            int idx = tid + u * 256;              // 1024 float4 units
            int r   = idx >> 3;                   // 8 units per row
            int kc  = (idx & 7) * 4;
            f32x4 v = *(const f32x4*)&A[(size_t)(m0 + r) * K + kt + kc];
            *(bf16x4*)&As[r][kc] = __builtin_convertvector(v, bf16x4);
        }
        // ---- stage B tile into [n][k] ----
        if (btrans == 0) {
#pragma unroll
            for (int u = 0; u < 2; ++u) {
                int idx = tid + u * 256;          // 512 float4 units along n
                int k   = idx >> 4;               // 16 units per k-row
                int nc  = (idx & 15) * 4;
                f32x4 v = *(const f32x4*)&B[(size_t)(kt + k) * Nn + n0 + nc];
                bf16x4 p = __builtin_convertvector(v, bf16x4);
                Bs[nc + 0][k] = p[0];
                Bs[nc + 1][k] = p[1];
                Bs[nc + 2][k] = p[2];
                Bs[nc + 3][k] = p[3];
            }
        } else {
#pragma unroll
            for (int u = 0; u < 2; ++u) {
                int idx = tid + u * 256;          // 512 float4 units along k
                int n   = idx >> 3;               // 8 units per n-row
                int kc  = (idx & 7) * 4;
                f32x4 v = *(const f32x4*)&B[(size_t)(n0 + n) * K + kt + kc];
                *(bf16x4*)&Bs[n][kc] = __builtin_convertvector(v, bf16x4);
            }
        }
        __syncthreads();

        // ---- fragments + 4 WMMAs (A-fragment reused across the 16x64 strip) ----
        const int arow = wv * 16 + lr;
        const int ka0 = half * 8;
        v8bf alo = *(const v8bf*)&As[arow][ka0];
        v8bf ahi = *(const v8bf*)&As[arow][ka0 + 16];
        v16bf afrag = mk_frag(alo, ahi);

#pragma unroll
        for (int t = 0; t < 4; ++t) {
            const int brow = t * 16 + lr;
            v8bf blo = *(const v8bf*)&Bs[brow][half * 8];
            v8bf bhi = *(const v8bf*)&Bs[brow][half * 8 + 16];
            v16bf bfrag = mk_frag(blo, bhi);
            v8f* accp = (t == 0) ? &acc0 : (t == 1) ? &acc1 : (t == 2) ? &acc2 : &acc3;
            *accp = __builtin_amdgcn_wmma_f32_16x16x32_bf16(
                false, afrag, false, bfrag, (short)0, *accp, false, false);
        }
        __syncthreads();
    }

    // ---- epilogue: C/D layout -> lane gives N, VGPR r gives M (lanes>=16: M+8) ----
    const int mrow = m0 + wv * 16 + half * 8;
#pragma unroll
    for (int t = 0; t < 4; ++t) {
        const v8f accv = (t == 0) ? acc0 : (t == 1) ? acc1 : (t == 2) ? acc2 : acc3;
        const int nn = n0 + t * 16 + lr;
        float bv = (mode == 1) ? bias[nn] : 0.0f;
#pragma unroll
        for (int r = 0; r < 8; ++r) {
            float v = accv[r];
            if (mode == 1) v += bv;
            else if (mode == 3) v = 1.0f / (1.0f + __expf(-v));
            C[(size_t)(mrow + r) * Nn + nn] = v;
        }
    }
}

// ---------------- host orchestration ----------------
static inline int cdiv(long long a, int b) { return (int)((a + b - 1) / b); }

extern "C" void kernel_launch(void* const* d_in, const int* in_sizes, int n_in,
                              void* d_out, int out_size, void* d_ws, size_t ws_size,
                              hipStream_t stream) {
    (void)in_sizes; (void)n_in; (void)out_size; (void)ws_size;

    const int*   x    = (const int*)d_in[0];
    const int*   eidx = (const int*)d_in[1];
    const int*   bat  = (const int*)d_in[2];
    const float* emb  = (const float*)d_in[3];
    const float* W1 = (const float*)d_in[4];  const float* b1 = (const float*)d_in[5];
    const float* W2 = (const float*)d_in[6];  const float* b2 = (const float*)d_in[7];
    const float* W3 = (const float*)d_in[8];  const float* b3 = (const float*)d_in[9];
    const float* W4 = (const float*)d_in[10]; const float* b4 = (const float*)d_in[11];
    const float* W5 = (const float*)d_in[12]; const float* b5 = (const float*)d_in[13];
    const float* Wd = (const float*)d_in[14]; const float* bd = (const float*)d_in[15];
    const int* src = eidx;
    const int* dst = eidx + GE;

    // output layout (concatenated, all float32)
    float* out        = (float*)d_out;
    float* out_z      = out;                                   // N*Dl
    float* out_ge     = out_z + (size_t)GN * GDL;              // B*Dl
    float* out_adj    = out_ge + (size_t)GB * GDL;             // B*NP*NP
    float* out_mask   = out_adj + (size_t)GB * GNP * GNP;      // B*NP
    float* out_logits = out_mask + (size_t)GB * GNP;           // N*T

    // workspace layout
    char* ws = (char*)d_ws;
    size_t off = 0;
    auto take = [&](size_t bytes) -> char* {
        char* p = ws + off;
        off = (off + bytes + 255) & ~(size_t)255;
        return p;
    };
    float*    buf0      = (float*)take((size_t)GN * GDH * 4);   // 32 MB panel A
    float*    buf1      = (float*)take((size_t)GN * GDH * 4);   // 32 MB panel B
    float*    z_dense   = (float*)take((size_t)GB * GNP * GDL * 4);
    float*    dis       = (float*)take((size_t)GN * 4);
    float*    self_norm = (float*)take((size_t)GN * 4);
    unsigned* indeg     = (unsigned*)take((size_t)GN * 4);
    unsigned* offsets   = (unsigned*)take((size_t)GN * 4);
    unsigned* cursor    = (unsigned*)take((size_t)GN * 4);
    unsigned* csr_src   = (unsigned*)take((size_t)GE * 4);
    float*    csr_w     = (float*)take((size_t)GE * 4);
    unsigned* counts    = (unsigned*)take((size_t)GB * 4);
    unsigned* starts    = (unsigned*)take((size_t)GB * 4);

    // ---- zero accumulators (deterministic across graph replays) ----
    zero_u32_kernel<<<cdiv(GN, 256), 256, 0, stream>>>(indeg, GN);
    zero_u32_kernel<<<cdiv(GN, 256), 256, 0, stream>>>(cursor, GN);
    zero_u32_kernel<<<1, 256, 0, stream>>>(counts, GB);
    zero_f32_kernel<<<cdiv((long long)GB * GNP * GDL, 256), 256, 0, stream>>>(z_dense, (long long)GB * GNP * GDL);
    zero_f32_kernel<<<cdiv((long long)GB * GNP, 256), 256, 0, stream>>>(out_mask, (long long)GB * GNP);
    zero_f32_kernel<<<cdiv((long long)GB * GDL, 256), 256, 0, stream>>>(out_ge, (long long)GB * GDL);

    // ---- degrees / norms / CSR ----
    count_u32_kernel<<<cdiv(GE, 256), 256, 0, stream>>>(dst, indeg, GE);
    node_norm_kernel<<<cdiv(GN, 256), 256, 0, stream>>>(indeg, dis, self_norm, GN);
    exclusive_scan_u32_kernel<<<1, 256, 0, stream>>>(indeg, offsets, GN);
    csr_fill_kernel<<<cdiv(GE, 256), 256, 0, stream>>>(src, dst, dis, offsets, cursor, csr_src, csr_w, GE);

    // ---- batch bookkeeping ----
    count_u32_kernel<<<cdiv(GN, 256), 256, 0, stream>>>(bat, counts, GN);
    exclusive_scan_u32_kernel<<<1, 256, 0, stream>>>(counts, starts, GB);

    // ---- embedding gather ----
    gather_emb_kernel<<<GN, GDE / 4, 0, stream>>>(x, emb, buf0);

    auto gemm = [&](const float* A, const float* Bm, const float* bias, float* C,
                    int M, int Nn, int K, int btrans, int mode, int batches,
                    long long sA, long long sB, long long sC) {
        dim3 g(M / 128, Nn / 64, batches);
        wmma_gemm_kernel<<<g, 256, 0, stream>>>(A, Bm, bias, C, M, Nn, K, btrans, mode, sA, sB, sC);
    };

    // ---- GCN layers (GEMM on matrix pipes, then CSR-gather aggregation) ----
    gemm(buf0, W1, nullptr, buf1, GN, GDH, GDE, 0, 0, 1, 0, 0, 0);
    gcn_aggregate_kernel<<<GN, GDH / 4, 0, stream>>>(buf1, buf0, csr_src, csr_w, offsets, indeg, self_norm, b1, GDH / 4, 1);

    gemm(buf0, W2, nullptr, buf1, GN, GDH, GDH, 0, 0, 1, 0, 0, 0);
    gcn_aggregate_kernel<<<GN, GDH / 4, 0, stream>>>(buf1, buf0, csr_src, csr_w, offsets, indeg, self_norm, b2, GDH / 4, 1);

    gemm(buf0, W3, nullptr, buf1, GN, GDH, GDH, 0, 0, 1, 0, 0, 0);
    gcn_aggregate_kernel<<<GN, GDH / 4, 0, stream>>>(buf1, buf0, csr_src, csr_w, offsets, indeg, self_norm, b3, GDH / 4, 1);

    gemm(buf0, W4, nullptr, buf1, GN, GDH, GDH, 0, 0, 1, 0, 0, 0);
    gcn_aggregate_kernel<<<GN, GDH / 4, 0, stream>>>(buf1, buf0, csr_src, csr_w, offsets, indeg, self_norm, b4, GDH / 4, 1);

    gemm(buf0, W5, nullptr, buf1, GN, GDL, GDH, 0, 0, 1, 0, 0, 0);
    gcn_aggregate_kernel<<<GN, GDL / 4, 0, stream>>>(buf1, out_z, csr_src, csr_w, offsets, indeg, self_norm, b5, GDL / 4, 0);

    // ---- graph mean pool ----
    pool_accum_kernel<<<GN, GDL, 0, stream>>>(out_z, bat, out_ge);
    pool_div_kernel<<<GB, GDL, 0, stream>>>(out_ge, counts);

    // ---- dense batch + mask ----
    scatter_dense_kernel<<<GN, GDL / 4, 0, stream>>>(out_z, bat, starts, z_dense, out_mask);

    // ---- pred_adj = sigmoid(Z Z^T), batched over B graphs ----
    gemm(z_dense, z_dense, nullptr, out_adj, GNP, GNP, GDL, /*btrans*/1, /*mode*/3, GB,
         (long long)GNP * GDL, (long long)GNP * GDL, (long long)GNP * GNP);

    // ---- node_logits = z @ Wd + bd ----
    gemm(out_z, Wd, bd, out_logits, GN, GT, GDL, 0, /*mode*/1, 1, 0, 0, 0);
}
